// DCTHighPass_23072564314729
// MI455X (gfx1250) — matrix operand order repaired and checked by reading, not compiled, not run on hardware
//
#include <hip/hip_runtime.h>
#include <math.h>

typedef __attribute__((ext_vector_type(16))) _Float16 v16h;
typedef __attribute__((ext_vector_type(8)))  float    v8f;

// cos(j*pi/4), j taken mod 8
__device__ __forceinline__ float qcos8(int j) {
    j &= 7;
    const float R = 0.70710678118654752440f;
    float v = 0.f;
    if (j == 0)                v =  1.f;
    else if (j == 4)           v = -1.f;
    else if (j == 1 || j == 7) v =  R;
    else if (j == 3 || j == 5) v = -R;
    return v;
}
__device__ __forceinline__ float qsin8(int j) { return qcos8(j + 6); }

// ---------------------------------------------------------------------------
// Kernel 1: gray -> per-8x8-tile FFT2 magnitude (via WMMA) -> mask -> ws
// Grid: 2048 blocks x 256 threads (8 waves). Each wave processes one pair of
// horizontally adjacent tiles per iteration; 8 iterations cover all
// 64*2048 = 131072 tile-pairs. All loop counts are block-uniform so
// __syncthreads() is safe and EXEC is all-ones at every WMMA.
// ---------------------------------------------------------------------------
__global__ __launch_bounds__(256) void fft_tiles_kernel(const float* __restrict__ x,
                                                        float* __restrict__ ws) {
    __shared__ float s_gray[8 * 128];  // per-wave 8x16 gray region
    __shared__ float s_mat [8 * 256];  // per-wave 16x16 f32 matrix buffer

    const int lane = threadIdx.x & 31;
    const int wave = threadIdx.x >> 5;
    float* gbuf = s_gray + wave * 128;
    float* mbuf = s_mat  + wave * 256;

    // ---- constant operands (built once per thread) ----
    // A1 = [[C,0],[0,-S]]  (16x32 f16 A-layout, K>=16 zero)
    // B2 = [[C^T,-S^T],[S^T,C^T]] = [[C,-S],[S,C]] (C,S symmetric), B-layout
    v16h a1, b2c;
    {
        const int m  = lane & 15;
        const int kh = lane >> 4;      // which K-half (0: K0..7, 1: K8..15)
#pragma unroll
        for (int i = 0; i < 16; ++i) { a1[i] = (_Float16)0.f; b2c[i] = (_Float16)0.f; }
#pragma unroll
        for (int idx = 0; idx < 8; ++idx) {
            float v = 0.f;
            if (m < 8 && kh == 0)       v =  qcos8(m * idx);
            else if (m >= 8 && kh == 1) v = -qsin8((m - 8) * idx);
            a1[idx] = (_Float16)v;
        }
        const int n = lane & 15;
        {
#pragma unroll
            for (int s = 0; s < 8; ++s) {
                const int k = kh * 8 + s;
                float v;
                if (k < 8) v = (n < 8) ?  qcos8(k * n)       : -qsin8(k * (n - 8));
                else       v = (n < 8) ?  qsin8((k - 8) * n) :  qcos8((k - 8) * (n - 8));
                b2c[s] = (_Float16)v;
            }
        }
    }

    const size_t chan = (size_t)512 * 512;

    for (int it = 0; it < 8; ++it) {
        const int p  = ((it * 2048 + (int)blockIdx.x) << 3) + wave; // pair id
        const int b  = p >> 11;        // batch
        const int pl = p & 2047;       // pair within image
        const int t0 = pl << 1;        // first tile index (i*64 + j)
        const int ti = t0 >> 6;        // tile row
        const int tj = t0 & 63;        // tile col (even)

        // ---- phase 1: gray for the 8x16 region (two tiles), coalesced ----
        {
            const int col = lane & 15;
            const int rh  = lane >> 4;
            const int xc  = tj * 8 + col;
#pragma unroll
            for (int q = 0; q < 4; ++q) {
                const int r = q * 2 + rh;
                const int y = ti * 8 + r;
                const size_t off = ((size_t)(b * 3) * 512 + y) * 512 + xc;
                const float R = x[off];
                const float G = x[off + chan];
                const float B = x[off + 2 * chan];
                gbuf[r * 16 + col] = 0.299f * R + 0.587f * G + 0.114f * B;
            }
        }
        __syncthreads();

        // ---- phase 2: B1 = [[T1 T2],[T1 T2]], WMMA1 -> D1 = [U_re; U_im] ----
        v8f d1 = {0.f, 0.f, 0.f, 0.f, 0.f, 0.f, 0.f, 0.f};
        {
            v16h b1;
            const int  n   = lane & 15;
#pragma unroll
            for (int r = 0; r < 8; ++r) {
                const float g = gbuf[r * 16 + n];
                const _Float16 h = (_Float16)g;
                b1[r]     = h;             // K rows 0..7 (lanes 0..15) / 8..15 (lanes 16..31)
                b1[r + 8] = (_Float16)0.f; // K rows 16..31 => zero
            }
            d1 = __builtin_amdgcn_wmma_f32_16x16x32_f16(false, a1, false, b1,
                                                        (short)0, d1, false, false);
        }
        {   // spill D1 (C/D layout: lane = column, vgpr = row) to 16x16 LDS
            const int colD  = lane & 15;
            const int rbase = (lane >> 4) * 8;
#pragma unroll
            for (int r = 0; r < 8; ++r) mbuf[(rbase + r) * 16 + colD] = d1[r];
        }
        __syncthreads();

        // ---- phase 3: A2 = [[Ure1,Uim1],[Ure2,Uim2]], WMMA2 -> D2 = [[Yre,Yim]x2]
        v8f d2 = {0.f, 0.f, 0.f, 0.f, 0.f, 0.f, 0.f, 0.f};
        {
            v16h a2;
            const int rowA = (lane & 7) + ((lane & 16) >> 1); // D1 source row
            const int coff = ((lane >> 3) & 1) * 8;           // D1 source col half
#pragma unroll
            for (int idx = 0; idx < 8; ++idx)
                a2[idx] = (_Float16)mbuf[rowA * 16 + coff + idx];
#pragma unroll
            for (int idx = 8; idx < 16; ++idx) a2[idx] = (_Float16)0.f;
            d2 = __builtin_amdgcn_wmma_f32_16x16x32_f16(false, a2, false, b2c,
                                                        (short)0, d2, false, false);
        }
        {
            const int colD  = lane & 15;
            const int rbase = (lane >> 4) * 8;
#pragma unroll
            for (int r = 0; r < 8; ++r) mbuf[(rbase + r) * 16 + colD] = d2[r];
        }
        __syncthreads();

        // ---- phase 4: |Y|, faithful mask bug, coalesced store to ws ----
        {
            const bool bmask = (b >= 3) && (b < 6);
#pragma unroll
            for (int k = 0; k < 4; ++k) {
                const int f  = k * 32 + lane;     // 128 outputs per tile-pair
                const int ts = f >> 6;            // which tile of the pair
                const int rr = (f >> 3) & 7;
                const int cc = f & 7;
                const float re = mbuf[(rr + 8 * ts) * 16 + cc];
                const float im = mbuf[(rr + 8 * ts) * 16 + cc + 8];
                float mag = sqrtf(re * re + im * im);
                if (bmask && rr >= 3 && rr < 6) mag = 0.f;
                ws[((size_t)(b * 4096 + t0 + ts) << 6) + rr * 8 + cc] = mag;
            }
        }
        __syncthreads();  // protect mbuf reads from next iteration's writes
    }
}

// ---------------------------------------------------------------------------
// Kernel 2: bilinear resize [64,1,32768,8] -> [64,1,512,512]
// Height scale = 64 => always rows 64*oy+31 / +32 with weight 0.5 each.
// Width: 8 -> 512 upsample with edge clamping. ws is L2-resident (67MB<192MB).
// ---------------------------------------------------------------------------
__global__ __launch_bounds__(256) void resize_kernel(const float* __restrict__ F,
                                                     float* __restrict__ out) {
    const int gid = blockIdx.x * 256 + threadIdx.x;
    const int ox = gid & 511;
    const int oy = (gid >> 9) & 511;
    const int b  = gid >> 18;

    float src = (ox + 0.5f) * 0.015625f - 0.5f;   // /64
    src = fminf(fmaxf(src, 0.f), 7.f);
    const int   x0 = (int)src;                    // src >= 0 so trunc == floor
    const int   x1 = min(x0 + 1, 7);
    const float wx = src - (float)x0;

    const size_t y0 = (size_t)64 * oy + 31;
    const float* rb = F + ((size_t)b << 18);      // b * 32768 * 8

    const float a0 = rb[y0 * 8 + x0];
    const float a1 = rb[y0 * 8 + x1];
    const float c0 = rb[(y0 + 1) * 8 + x0];
    const float c1 = rb[(y0 + 1) * 8 + x1];
    const float r0 = a0 * 0.5f + c0 * 0.5f;       // height lerp (w = 0.5)
    const float r1 = a1 * 0.5f + c1 * 0.5f;
    out[gid] = r0 * (1.f - wx) + r1 * wx;         // width lerp, reference order
}

extern "C" void kernel_launch(void* const* d_in, const int* in_sizes, int n_in,
                              void* d_out, int out_size, void* d_ws, size_t ws_size,
                              hipStream_t stream) {
    (void)in_sizes; (void)n_in; (void)out_size; (void)ws_size;
    const float* x  = (const float*)d_in[0];
    float* out      = (float*)d_out;
    float* ws       = (float*)d_ws;   // needs 64*4096*64*4 = 64 MiB

    fft_tiles_kernel<<<2048, 256, 0, stream>>>(x, ws);
    resize_kernel<<<65536, 256, 0, stream>>>(ws, out);
}
